// MDTBlock_3865470566442
// MI455X (gfx1250) — matrix operand order repaired and checked by reading, not compiled, not run on hardware
//
#include <hip/hip_runtime.h>

#define DEV __device__ __forceinline__

typedef __attribute__((ext_vector_type(16))) _Float16 v16h;
typedef __attribute__((ext_vector_type(8)))  float    v8f;

#define WMMA(a,b,c) __builtin_amdgcn_wmma_f32_16x16x32_f16(false,(a),false,(b),(short)0,(c),false,false)

#if __has_builtin(__builtin_amdgcn_global_load_async_to_lds_b128)
#define HAVE_ASYNC 1
typedef int v4i_ty __attribute__((vector_size(16)));
typedef __attribute__((address_space(1))) v4i_ty* gas_v4i;
typedef __attribute__((address_space(3))) v4i_ty* las_v4i;
DEV void async_cp16(const _Float16* g, _Float16* l) {
    // generic->addrspace via integer cast; LDS generic addresses truncate to the
    // 32-bit LDS offset per the gfx1250 aperture rules.
    __builtin_amdgcn_global_load_async_to_lds_b128(
        (gas_v4i)(size_t)g, (las_v4i)(unsigned)(size_t)l, 0, 0);
}
DEV void wait_async0() {
#if __has_builtin(__builtin_amdgcn_s_wait_asynccnt)
    __builtin_amdgcn_s_wait_asynccnt(0);
#else
    asm volatile("s_wait_asynccnt 0" ::: "memory");
#endif
}
#else
#define HAVE_ASYNC 0
#endif

// ---------------- fragment loaders (CDNA5 WMMA layouts) ----------------
// A matrix 16x32 f16: lane row = lane&15; K halves [k0, k0+8) and [k0+16, k0+24), k0 = (lane>>4)*8
DEV v16h frag_ld_a(const _Float16* p, int stride) {
    const int lane = threadIdx.x & 31;
    const int row  = lane & 15;
    const int k0   = (lane >> 4) << 3;
    const _Float16* q = p + (size_t)row * stride + k0;
    union { v16h v; uint4 u[2]; } r;
    r.u[0] = *(const uint4*)(q);
    r.u[1] = *(const uint4*)(q + 16);
    return r.v;
}
// B matrix 32x16 f16 sourced from B^T stored row-major (n rows, k cols):
// lane col n = lane&15; K halves [k0, k0+8) and [k0+16, k0+24), k0 = (lane>>4)*8
DEV v16h frag_ld_b(const _Float16* p, int stride) {
    const int lane = threadIdx.x & 31;
    const int n  = lane & 15;
    const int k0 = (lane >> 4) << 3;
    const _Float16* q = p + (size_t)n * stride + k0;
    union { v16h v; uint4 u[2]; } r;
    r.u[0] = *(const uint4*)(q);
    r.u[1] = *(const uint4*)(q + 16);
    return r.v;
}
// strided-element variants for the head-interleaved (B,N,DH,H) layout
DEV v16h frag_ld_a_strided(const _Float16* p, int rstride, int estride) {
    const int lane = threadIdx.x & 31;
    const int row  = lane & 15;
    const int k0   = (lane >> 4) << 3;
    const _Float16* q = p + (size_t)row * rstride + (size_t)k0 * estride;
    v16h r;
#pragma unroll
    for (int i = 0; i < 8; ++i) r[i] = q[i * estride];
#pragma unroll
    for (int i = 0; i < 8; ++i) r[8 + i] = q[(16 + i) * estride];
    return r;
}
DEV v16h frag_ld_b_strided(const _Float16* p, int rstride, int estride) {
    const int lane = threadIdx.x & 31;
    const int n  = lane & 15;
    const int k0 = (lane >> 4) << 3;
    const _Float16* q = p + (size_t)n * rstride + (size_t)k0 * estride;
    v16h r;
#pragma unroll
    for (int i = 0; i < 8; ++i) r[i] = q[i * estride];
#pragma unroll
    for (int i = 0; i < 8; ++i) r[8 + i] = q[(16 + i) * estride];
    return r;
}

DEV v8f vzero8() { v8f z = {0.f,0.f,0.f,0.f,0.f,0.f,0.f,0.f}; return z; }

DEV float xormax16(float v) {
    v = fmaxf(v, __shfl_xor(v, 1, 32));
    v = fmaxf(v, __shfl_xor(v, 2, 32));
    v = fmaxf(v, __shfl_xor(v, 4, 32));
    v = fmaxf(v, __shfl_xor(v, 8, 32));
    return v;
}
DEV float xorsum16(float v) {
    v += __shfl_xor(v, 1, 32);
    v += __shfl_xor(v, 2, 32);
    v += __shfl_xor(v, 4, 32);
    v += __shfl_xor(v, 8, 32);
    return v;
}

DEV float gelu_tanh(float x) {
    float x3 = x * x * x;
    float t  = tanhf(0.7978845608028654f * (x + 0.044715f * x3));
    return 0.5f * x * (1.0f + t);
}

// ---------------- small prep kernels ----------------

// transpose + downconvert weights: in (K x Nn) f32 row-major -> out (Nn x K) f16 row-major
__global__ void __launch_bounds__(256) wtrans_kernel(const float* __restrict__ in,
                                                     _Float16* __restrict__ out,
                                                     int K, int Nn) {
    size_t i = (size_t)blockIdx.x * 256 + threadIdx.x;
    if (i < (size_t)K * Nn) {
        int k = (int)(i / Nn), n = (int)(i % Nn);
        out[(size_t)n * K + k] = (_Float16)in[i];
    }
}

__global__ void __launch_bounds__(256) tof16_kernel(const float* __restrict__ in,
                                                    _Float16* __restrict__ out, int n) {
    int i = blockIdx.x * 256 + threadIdx.x;
    if (i < n) out[i] = (_Float16)in[i];
}

// ada = silu(noise) @ W_ada + b_ada   (one block per batch row)
__global__ void __launch_bounds__(256) ada_kernel(const float* __restrict__ noise,
                                                  const float* __restrict__ W_ada,
                                                  const float* __restrict__ b_ada,
                                                  float* __restrict__ ada) {
    const int b = blockIdx.x;
    __shared__ float ss[1024];
    for (int i = threadIdx.x; i < 1024; i += 256) {
        float v = noise[(size_t)b * 1024 + i];
        ss[i] = v / (1.f + __expf(-v));
    }
    __syncthreads();
    for (int j0 = 0; j0 < 6144; j0 += 256) {
        int j = j0 + threadIdx.x;
        float acc = b_ada[j];
        for (int c = 0; c < 1024; ++c) acc = fmaf(ss[c], W_ada[(size_t)c * 6144 + j], acc);
        ada[(size_t)b * 6144 + j] = acc;
    }
}

// LayerNorm + adaLN modulate -> f16 (one block per token)
__global__ void __launch_bounds__(256) ln_mod_kernel(const float* __restrict__ x,
                                                     const float* __restrict__ ada,
                                                     int shOff, int scOff,
                                                     _Float16* __restrict__ out) {
    const int tok = blockIdx.x;
    const int b   = tok >> 8;
    const float* xr = x + (size_t)tok * 1024;
    float lx[4], s = 0.f, s2 = 0.f;
#pragma unroll
    for (int i = 0; i < 4; ++i) {
        lx[i] = xr[threadIdx.x + i * 256];
        s  += lx[i];
        s2 += lx[i] * lx[i];
    }
    __shared__ float rA[256], rB[256];
    rA[threadIdx.x] = s; rB[threadIdx.x] = s2;
    __syncthreads();
    for (int st = 128; st > 0; st >>= 1) {
        if ((int)threadIdx.x < st) { rA[threadIdx.x] += rA[threadIdx.x + st]; rB[threadIdx.x] += rB[threadIdx.x + st]; }
        __syncthreads();
    }
    const float mean = rA[0] * (1.f / 1024.f);
    const float var  = rB[0] * (1.f / 1024.f) - mean * mean;
    const float rstd = rsqrtf(var + 1e-6f);
    const float* sh = ada + (size_t)b * 6144 + shOff;
    const float* sc = ada + (size_t)b * 6144 + scOff;
#pragma unroll
    for (int i = 0; i < 4; ++i) {
        int c = threadIdx.x + i * 256;
        float h = (lx[i] - mean) * rstd * (1.f + sc[c]) + sh[c];
        out[(size_t)tok * 1024 + c] = (_Float16)h;
    }
}

// ---------------- WMMA GEMM: D = A(MxK f16) * Bt(NxK f16)^T  ----------------
// EPI 0: out f16 = acc + bias ; EPI 1: out f16 = gelu(acc + bias)
// EPI 2: out f32 = resid + gate[b,col] * (acc + bias)
template <int EPI>
__global__ void __launch_bounds__(256) gemm_kernel(const _Float16* __restrict__ A,
                                                   const _Float16* __restrict__ Bt,
                                                   const float* __restrict__ bias,
                                                   const float* __restrict__ resid,
                                                   const float* __restrict__ gate,
                                                   float* __restrict__ outF,
                                                   _Float16* __restrict__ outH,
                                                   int Mtot, int Nout, int K) {
    __shared__ _Float16 sA[128 * 32];
    __shared__ _Float16 sB[64 * 32];
    const int m0 = blockIdx.y * 128;
    const int n0 = blockIdx.x * 64;
    const int wave = threadIdx.x >> 5;
    const int lane = threadIdx.x & 31;
    const int li = lane & 15, hi = lane >> 4;

    v8f acc[4];
#pragma unroll
    for (int nt = 0; nt < 4; ++nt) acc[nt] = vzero8();

    const int ar  = threadIdx.x >> 1;            // A tile row
    const int aof = (threadIdx.x & 1) << 4;      // half-offset within 32-k row
    const int gm  = m0 + ar;
    const int br  = threadIdx.x >> 2;            // B tile row (n)
    const int bof = (threadIdx.x & 3) << 3;

    for (int k0 = 0; k0 < K; k0 += 32) {
        __syncthreads();
#if HAVE_ASYNC
        // direct global->LDS via the async data path (ASYNCcnt tracked).
        // OOB A rows (kv GEMM tail) read padded workspace; stores are row-guarded.
        async_cp16(A + (size_t)gm * K + k0 + aof,     sA + ar * 32 + aof);
        async_cp16(A + (size_t)gm * K + k0 + aof + 8, sA + ar * 32 + aof + 8);
        async_cp16(Bt + (size_t)(n0 + br) * K + k0 + bof, sB + br * 32 + bof);
        wait_async0();
#else
        uint4 a0 = {0,0,0,0}, a1 = {0,0,0,0};
        if (gm < Mtot) {
            const uint4* src = (const uint4*)(A + (size_t)gm * K + k0 + aof);
            a0 = src[0]; a1 = src[1];
            if (k0 + 32 < K) __builtin_prefetch(A + (size_t)gm * K + k0 + 32 + aof, 0, 1);
        }
        *(uint4*)(sA + ar * 32 + aof)     = a0;
        *(uint4*)(sA + ar * 32 + aof + 8) = a1;
        *(uint4*)(sB + br * 32 + bof) = *(const uint4*)(Bt + (size_t)(n0 + br) * K + k0 + bof);
#endif
        __syncthreads();

        v16h af = frag_ld_a(sA + wave * 16 * 32, 32);
#pragma unroll
        for (int nt = 0; nt < 4; ++nt) {
            v16h bf = frag_ld_b(sB + nt * 16 * 32, 32);
            acc[nt] = WMMA(af, bf, acc[nt]);
        }
    }

#pragma unroll
    for (int nt = 0; nt < 4; ++nt) {
        const int col = n0 + nt * 16 + li;
        const float bv = bias ? bias[col] : 0.f;
#pragma unroll
        for (int v = 0; v < 8; ++v) {
            const int row = m0 + wave * 16 + v + 8 * hi;
            if (row < Mtot) {
                float xv = acc[nt][v] + bv;
                const size_t idx = (size_t)row * Nout + col;
                if (EPI == 1) xv = gelu_tanh(xv);
                if (EPI == 2) {
                    const int bb = row >> 8;
                    outF[idx] = resid[idx] + gate[(size_t)bb * 6144 + col] * xv;
                } else {
                    outH[idx] = (_Float16)xv;
                }
            }
        }
    }
}

// ---------------- fused self-attention with relative-position bias ----------------
// grid = B*H blocks, 256 threads (8 waves x 32 query rows). qkv: (8192 x 3072) f16,
// channel = s*1024 + h*64 + d. o16: (8192 x 1024) f16, channel = h*64 + d.
__global__ void __launch_bounds__(256) attn_self_kernel(const _Float16* __restrict__ qkv,
                                                        const float* __restrict__ rel_table,
                                                        _Float16* __restrict__ o16) {
    const int bh = blockIdx.x, b = bh >> 4, h = bh & 15;
    const int wave = threadIdx.x >> 5, lane = threadIdx.x & 31;
    const int li = lane & 15, hi = lane >> 4;

    __shared__ _Float16 sVt[64 * 32];      // V^T chunk: [d][key]
    __shared__ _Float16 sP[8][16 * 32];    // per-wave P staging
    __shared__ float    sBias[961];        // rel_table[:,h]

    for (int i = threadIdx.x; i < 961; i += 256) sBias[i] = rel_table[(size_t)i * 16 + h];

    const size_t base = (size_t)b * 256 * 3072 + (size_t)h * 64;
    const _Float16* Q = qkv + base;
    const _Float16* K = qkv + base + 1024;
    const _Float16* V = qkv + base + 2048;
    const int row0 = wave * 32;

    v16h qf[2][2];
#pragma unroll
    for (int mt = 0; mt < 2; ++mt)
#pragma unroll
        for (int ks = 0; ks < 2; ++ks)
            qf[mt][ks] = frag_ld_a(Q + (size_t)(row0 + mt * 16) * 3072 + ks * 32, 3072);

    v8f accO[2][4];
    float rm[2][8], rs[2][8];
#pragma unroll
    for (int mt = 0; mt < 2; ++mt) {
#pragma unroll
        for (int dt = 0; dt < 4; ++dt) accO[mt][dt] = vzero8();
#pragma unroll
        for (int v = 0; v < 8; ++v) { rm[mt][v] = -1e30f; rs[mt][v] = 0.f; }
    }
    const float scale = 0.125f;  // DH^-0.5

    for (int c = 0; c < 8; ++c) {
        const int key0 = c * 32;
        __syncthreads();
        for (int i = threadIdx.x; i < 2048; i += 256) {
            const int d = i >> 5, kk = i & 31;
            sVt[d * 32 + kk] = V[(size_t)(key0 + kk) * 3072 + d];
        }
        __syncthreads();

#pragma unroll
        for (int mt = 0; mt < 2; ++mt) {
            v8f s[2];
#pragma unroll
            for (int nt = 0; nt < 2; ++nt) {
                v8f z = vzero8();
#pragma unroll
                for (int ks = 0; ks < 2; ++ks) {
                    v16h kf = frag_ld_b(K + (size_t)(key0 + nt * 16) * 3072 + ks * 32, 3072);
                    z = WMMA(qf[mt][ks], kf, z);
                }
                s[nt] = z;
            }
            // scale + relative position bias (window geometry: n=(i,j), 16x16)
#pragma unroll
            for (int nt = 0; nt < 2; ++nt) {
                const int col = key0 + nt * 16 + li;
                const int i2 = col >> 4, j2 = col & 15;
#pragma unroll
                for (int v = 0; v < 8; ++v) {
                    const int row = row0 + mt * 16 + v + 8 * hi;
                    const int i1 = row >> 4, j1 = row & 15;
                    const int idx = (i1 - i2 + 15) * 31 + (j1 - j2 + 15);
                    s[nt][v] = s[nt][v] * scale + sBias[idx];
                }
            }
            // online softmax
#pragma unroll
            for (int v = 0; v < 8; ++v) {
                float cm = xormax16(fmaxf(s[0][v], s[1][v]));
                const float mn = fmaxf(rm[mt][v], cm);
                const float alpha = __expf(rm[mt][v] - mn);
                rm[mt][v] = mn;
                const float p0 = __expf(s[0][v] - mn);
                const float p1 = __expf(s[1][v] - mn);
                s[0][v] = p0; s[1][v] = p1;
                rs[mt][v] = rs[mt][v] * alpha + xorsum16(p0 + p1);
#pragma unroll
                for (int dt = 0; dt < 4; ++dt) accO[mt][dt][v] *= alpha;
            }
            // stage P -> LDS -> A fragment, then P @ V
            _Float16* ps = sP[wave];
#pragma unroll
            for (int nt = 0; nt < 2; ++nt)
#pragma unroll
                for (int v = 0; v < 8; ++v)
                    ps[(v + 8 * hi) * 32 + nt * 16 + li] = (_Float16)s[nt][v];
            asm volatile("s_wait_dscnt 0" ::: "memory");
            v16h pf = frag_ld_a(ps, 32);
#pragma unroll
            for (int dt = 0; dt < 4; ++dt) {
                v16h vf = frag_ld_b(sVt + dt * 16 * 32, 32);
                accO[mt][dt] = WMMA(pf, vf, accO[mt][dt]);
            }
        }
    }
    // normalize + store (channel = h*64 + d)
#pragma unroll
    for (int mt = 0; mt < 2; ++mt)
#pragma unroll
        for (int v = 0; v < 8; ++v) {
            const float inv = 1.f / rs[mt][v];
            const int row = row0 + mt * 16 + v + 8 * hi;
            const size_t ob = (size_t)(b * 256 + row) * 1024 + h * 64;
#pragma unroll
            for (int dt = 0; dt < 4; ++dt)
                o16[ob + dt * 16 + li] = (_Float16)(accO[mt][dt][v] * inv);
        }
}

// ---------------- fused cross-attention, head-interleaved (B,N,DH,H) ----------------
// q2: 8192x1024 f16, channel(d,h) = d*16+h ; kv: padded (2464+32)x2048 f16, k at d*16+h, v at 1024+d*16+h
// x (f32, 8192x1024) updated in place: x += softmax(q2 k2^T / 8) v2
__global__ void __launch_bounds__(256) attn_cross_kernel(const _Float16* __restrict__ q2,
                                                         const _Float16* __restrict__ kv,
                                                         float* __restrict__ x) {
    const int bh = blockIdx.x, b = bh >> 4, h = bh & 15;
    const int wave = threadIdx.x >> 5, lane = threadIdx.x & 31;
    const int li = lane & 15, hi = lane >> 4;

    __shared__ _Float16 sVt[64 * 32];
    __shared__ _Float16 sP[8][16 * 32];

    const _Float16* Q  = q2 + (size_t)b * 256 * 1024 + h;
    const _Float16* Kp = kv + (size_t)b * 77 * 2048 + h;
    const _Float16* Vp = kv + (size_t)b * 77 * 2048 + 1024 + h;
    const int row0 = wave * 32;

    v16h qf[2][2];
#pragma unroll
    for (int mt = 0; mt < 2; ++mt)
#pragma unroll
        for (int ks = 0; ks < 2; ++ks)
            qf[mt][ks] = frag_ld_a_strided(Q + (size_t)(row0 + mt * 16) * 1024 + ks * 512, 1024, 16);

    v8f accO[2][4];
    float rm[2][8], rs[2][8];
#pragma unroll
    for (int mt = 0; mt < 2; ++mt) {
#pragma unroll
        for (int dt = 0; dt < 4; ++dt) accO[mt][dt] = vzero8();
#pragma unroll
        for (int v = 0; v < 8; ++v) { rm[mt][v] = -1e30f; rs[mt][v] = 0.f; }
    }

    for (int c = 0; c < 3; ++c) {          // 3 chunks of 32 cover 77 (padded to 96)
        const int key0 = c * 32;
        __syncthreads();
        for (int i = threadIdx.x; i < 2048; i += 256) {
            const int d = i >> 5, kk = i & 31;
            const int key = key0 + kk;
            sVt[d * 32 + kk] = (key < 77) ? Vp[(size_t)key * 2048 + d * 16] : (_Float16)0.f;
        }
        __syncthreads();

#pragma unroll
        for (int mt = 0; mt < 2; ++mt) {
            v8f s[2];
#pragma unroll
            for (int nt = 0; nt < 2; ++nt) {
                v8f z = vzero8();
#pragma unroll
                for (int ks = 0; ks < 2; ++ks) {
                    v16h kf = frag_ld_b_strided(Kp + (size_t)(key0 + nt * 16) * 2048 + ks * 512, 2048, 16);
                    z = WMMA(qf[mt][ks], kf, z);
                }
                s[nt] = z;
            }
#pragma unroll
            for (int nt = 0; nt < 2; ++nt) {
                const int col = key0 + nt * 16 + li;
#pragma unroll
                for (int v = 0; v < 8; ++v)
                    s[nt][v] = (col < 77) ? s[nt][v] * 0.125f : -1e30f;
            }
#pragma unroll
            for (int v = 0; v < 8; ++v) {
                float cm = xormax16(fmaxf(s[0][v], s[1][v]));
                const float mn = fmaxf(rm[mt][v], cm);
                const float alpha = __expf(rm[mt][v] - mn);
                rm[mt][v] = mn;
                const float p0 = __expf(s[0][v] - mn);
                const float p1 = __expf(s[1][v] - mn);
                s[0][v] = p0; s[1][v] = p1;
                rs[mt][v] = rs[mt][v] * alpha + xorsum16(p0 + p1);
#pragma unroll
                for (int dt = 0; dt < 4; ++dt) accO[mt][dt][v] *= alpha;
            }
            _Float16* ps = sP[wave];
#pragma unroll
            for (int nt = 0; nt < 2; ++nt)
#pragma unroll
                for (int v = 0; v < 8; ++v)
                    ps[(v + 8 * hi) * 32 + nt * 16 + li] = (_Float16)s[nt][v];
            asm volatile("s_wait_dscnt 0" ::: "memory");
            v16h pf = frag_ld_a(ps, 32);
#pragma unroll
            for (int dt = 0; dt < 4; ++dt) {
                v16h vf = frag_ld_b(sVt + dt * 16 * 32, 32);
                accO[mt][dt] = WMMA(pf, vf, accO[mt][dt]);
            }
        }
    }
    // x += O / l  at channel d*16+h
#pragma unroll
    for (int mt = 0; mt < 2; ++mt)
#pragma unroll
        for (int v = 0; v < 8; ++v) {
            const float inv = 1.f / rs[mt][v];
            const int row = row0 + mt * 16 + v + 8 * hi;
            const size_t tb = (size_t)(b * 256 + row) * 1024;
#pragma unroll
            for (int dt = 0; dt < 4; ++dt) {
                const int d = dt * 16 + li;
                x[tb + d * 16 + h] += accO[mt][dt][v] * inv;
            }
        }
}

// ---------------- host orchestration ----------------
static inline size_t align256(size_t v) { return (v + 255) & ~(size_t)255; }

extern "C" void kernel_launch(void* const* d_in, const int* in_sizes, int n_in,
                              void* d_out, int out_size, void* d_ws, size_t ws_size,
                              hipStream_t stream) {
    const float* x      = (const float*)d_in[0];
    const float* noise  = (const float*)d_in[1];
    const float* y      = (const float*)d_in[2];
    const float* W_qkv  = (const float*)d_in[3];
    const float* b_qkv  = (const float*)d_in[4];
    const float* W_proj = (const float*)d_in[5];
    const float* b_proj = (const float*)d_in[6];
    const float* rel    = (const float*)d_in[7];
    const float* W_ada  = (const float*)d_in[8];
    const float* b_ada  = (const float*)d_in[9];
    const float* W_fc1  = (const float*)d_in[10];
    const float* b_fc1  = (const float*)d_in[11];
    const float* W_fc2  = (const float*)d_in[12];
    const float* b_fc2  = (const float*)d_in[13];
    const float* W_q    = (const float*)d_in[14];
    const float* W_kv   = (const float*)d_in[15];
    float* out = (float*)d_out;

    char* ws = (char*)d_ws;
    size_t off = 0;
    auto alloc = [&](size_t bytes) { size_t o = off; off = align256(off + bytes); return o; };

    float*    ada    = (float*)(ws + alloc((size_t)32 * 6144 * 4));
    _Float16* WqkvT  = (_Float16*)(ws + alloc((size_t)3072 * 1024 * 2));
    _Float16* WprojT = (_Float16*)(ws + alloc((size_t)1024 * 1024 * 2));
    _Float16* WqT    = (_Float16*)(ws + alloc((size_t)1024 * 1024 * 2));
    _Float16* WkvT   = (_Float16*)(ws + alloc((size_t)2048 * 1024 * 2));
    _Float16* Wfc1T  = (_Float16*)(ws + alloc((size_t)4096 * 1024 * 2));
    _Float16* Wfc2T  = (_Float16*)(ws + alloc((size_t)1024 * 4096 * 2));
    _Float16* h16    = (_Float16*)(ws + alloc((size_t)8192 * 1024 * 2));
    _Float16* bigH   = (_Float16*)(ws + alloc((size_t)8192 * 4096 * 2));  // overlapped region
    float*    x1     = (float*)(ws + alloc((size_t)8192 * 1024 * 4));
    (void)ws_size; (void)in_sizes; (void)n_in; (void)out_size;

    // overlapped sub-buffers of bigH (lifetimes are disjoint)
    _Float16* qkv16 = bigH;                                   // 8192x3072
    _Float16* x1_16 = bigH;                                   // 8192x1024
    _Float16* q2_16 = bigH + (size_t)8192 * 1024;             // 8192x1024
    _Float16* y16   = bigH + (size_t)2 * 8192 * 1024;         // 2464x1024 (pad to 2560 rows)
    _Float16* kv16  = bigH + (size_t)2 * 8192 * 1024 + (size_t)2560 * 1024;  // 2496x2048 (padded rows)
    _Float16* h1_16 = bigH;                                   // 8192x4096
    _Float16* o16   = h16;                                    // reuse LN buffer for attention out

    // 1. weight transpose + f16 downconvert
    auto WT = [&](const float* src, _Float16* dst, int K, int Nn) {
        size_t tot = (size_t)K * Nn;
        wtrans_kernel<<<dim3((unsigned)((tot + 255) / 256)), 256, 0, stream>>>(src, dst, K, Nn);
    };
    WT(W_qkv, WqkvT, 1024, 3072);
    WT(W_proj, WprojT, 1024, 1024);
    WT(W_q, WqT, 1024, 1024);
    WT(W_kv, WkvT, 1024, 2048);
    WT(W_fc1, Wfc1T, 1024, 4096);
    WT(W_fc2, Wfc2T, 4096, 1024);

    // 2. adaLN vector
    ada_kernel<<<32, 256, 0, stream>>>(noise, W_ada, b_ada, ada);

    // 3. LN + modulate (MSA) -> h16
    ln_mod_kernel<<<8192, 256, 0, stream>>>(x, ada, 0, 1024, h16);

    // 4. qkv = h @ W_qkv + b
    gemm_kernel<0><<<dim3(48, 64), 256, 0, stream>>>(h16, WqkvT, b_qkv, nullptr, nullptr,
                                                     nullptr, qkv16, 8192, 3072, 1024);
    // 5. windowed self-attention with relative bias
    attn_self_kernel<<<512, 256, 0, stream>>>(qkv16, rel, o16);

    // 6. x1 = x + g_msa * (o @ W_proj + b)
    gemm_kernel<2><<<dim3(16, 64), 256, 0, stream>>>(o16, WprojT, b_proj, x, ada + 2 * 1024,
                                                     x1, nullptr, 8192, 1024, 1024);
    // 7-10. cross-attention inputs
    tof16_kernel<<<32768, 256, 0, stream>>>(x1, x1_16, 8192 * 1024);
    gemm_kernel<0><<<dim3(16, 64), 256, 0, stream>>>(x1_16, WqT, nullptr, nullptr, nullptr,
                                                     nullptr, q2_16, 8192, 1024, 1024);
    tof16_kernel<<<(2523136 + 255) / 256, 256, 0, stream>>>(y, y16, 2523136);
    gemm_kernel<0><<<dim3(32, 20), 256, 0, stream>>>(y16, WkvT, nullptr, nullptr, nullptr,
                                                     nullptr, kv16, 2464, 2048, 1024);
    // 11. x1 += cross-attn(q2, kv)   (in place -> x2)
    attn_cross_kernel<<<512, 256, 0, stream>>>(q2_16, kv16, x1);

    // 12. LN + modulate (MLP) -> h16
    ln_mod_kernel<<<8192, 256, 0, stream>>>(x1, ada, 3 * 1024, 4 * 1024, h16);

    // 13. h1 = gelu(h @ W_fc1 + b)
    gemm_kernel<1><<<dim3(64, 64), 256, 0, stream>>>(h16, Wfc1T, b_fc1, nullptr, nullptr,
                                                     nullptr, h1_16, 8192, 4096, 1024);
    // 14. out = x2 + g_mlp * (h1 @ W_fc2 + b)
    gemm_kernel<2><<<dim3(16, 64), 256, 0, stream>>>(h1_16, Wfc2T, b_fc2, x1, ada + 5 * 1024,
                                                     out, nullptr, 8192, 1024, 4096);
}
